// CleanupBlock_7851200217999
// MI455X (gfx1250) — compile-verified, matrix operand order checked by reading, not run on hardware
//
#include <hip/hip_runtime.h>
#include <hip/hip_bf16.h>

typedef __attribute__((ext_vector_type(16))) _Float16 v16h;
typedef __attribute__((ext_vector_type(8)))  _Float16 v8h;
typedef __attribute__((ext_vector_type(4)))  _Float16 v4h;
typedef __attribute__((ext_vector_type(2)))  _Float16 v2h;
typedef __attribute__((ext_vector_type(8)))  float    v8f;
typedef __attribute__((ext_vector_type(4)))  float    v4f;
typedef __attribute__((ext_vector_type(4)))  int      v4i;

typedef v4i __attribute__((address_space(1)))* gptr128;   // global int4*
typedef v4i __attribute__((address_space(3)))* lptr128;   // LDS int4*

#define MD        1024          // model dims (K of GEMM1, N of GEMM2)
#define NS        256           // supports   (N of GEMM1, K of GEMM2)
#define NLAYERS   4
#define ROWS_TOT  (4 * 4096)
#define WPB       4             // waves per block
#define RPW       16            // rows per wave (WMMA M tile)
#define HSTRIDE   1032          // f16 elems, multiple of 8
#define WSTRIDE   264           // f16 elems, multiple of 8

#define KC        64            // GEMM1 K-chunk staged in LDS
#define KSTR      72            // keys stage row stride (bank-spread, mult of 8)
#define DC        64            // GEMM2 d-chunk staged in LDS (4 col-blocks)
#define VSTR      264           // values stage row stride
#define NKCH      (MD / KC)     // 16 chunks
#define NDCH      (MD / DC)     // 16 chunks

#if __has_builtin(__builtin_amdgcn_global_load_async_to_lds_b128)
#define HAVE_ASYNC_LDS 1
#else
#define HAVE_ASYNC_LDS 0
#endif

// 16B global -> LDS copy: async (ASYNCcnt) if available, else direct.
__device__ __forceinline__ void copy16_to_lds(const _Float16* g, _Float16* l) {
#if HAVE_ASYNC_LDS
  __builtin_amdgcn_global_load_async_to_lds_b128(
      (gptr128)g, (lptr128)l, 0, 0);
#else
  *(v8h*)l = *(const v8h*)g;
#endif
}

__device__ __forceinline__ void staging_wait() {
#if HAVE_ASYNC_LDS
#if __has_builtin(__builtin_amdgcn_s_wait_asynccnt)
  __builtin_amdgcn_s_wait_asynccnt(0);
#else
  asm volatile("s_wait_asynccnt 0x0" ::: "memory");
#endif
#endif
}

__device__ __forceinline__ void sched_fence() {
  __builtin_amdgcn_sched_barrier(0);
}

__device__ __forceinline__ v16h make16(v8h lo, v8h hi) {
  v16h r;
  #pragma unroll
  for (int i = 0; i < 8; ++i) { r[i] = lo[i]; r[8 + i] = hi[i]; }
  return r;
}
__device__ __forceinline__ v16h load_frag(const _Float16* p) {
  return make16(*(const v8h*)p, *(const v8h*)(p + 16));
}

#define WMMA(A, B, C)                                                        \
  __builtin_amdgcn_wmma_f32_16x16x32_f16(false, (A), false, (B), (short)0,   \
                                         (C), false, false)

// ---------------------------------------------------------------------------
// Prep: keys fp32 -> f16 same layout; values fp32 -> f16 transposed [L][MD][NS]
// ---------------------------------------------------------------------------
__global__ void cleanup_prep_kv(const float* __restrict__ keys,
                                const float* __restrict__ values,
                                _Float16* __restrict__ kf,
                                _Float16* __restrict__ vtf) {
  int idx = blockIdx.x * blockDim.x + threadIdx.x;
  const int total = NLAYERS * NS * MD;
  if (idx < total) {
    kf[idx] = (_Float16)keys[idx];
    int l = idx / (NS * MD);
    int r = idx - l * (NS * MD);
    int n = r / MD;
    int d = r - n * MD;
    vtf[(size_t)l * (MD * NS) + (size_t)d * NS + n] = (_Float16)values[idx];
  }
}

// ---------------------------------------------------------------------------
// Fused 4-layer chain; async-staged double-buffered K/V; WMMA f16 -> f32.
// ---------------------------------------------------------------------------
__global__ __launch_bounds__(WPB * 32, 1)
__attribute__((amdgpu_waves_per_eu(1)))
void cleanup_chain_wmma(const float* __restrict__ x,
                        const _Float16* __restrict__ kf,
                        const _Float16* __restrict__ vtf,
                        float* __restrict__ out) {
  __shared__ _Float16 Hs[WPB][RPW][HSTRIDE];     // 132,096 B
  __shared__ _Float16 Ws[WPB][RPW][WSTRIDE];     //  33,792 B
  __shared__ _Float16 Stage[2][NS * KSTR];       //  73,728 B (V view: 64*264 fits)

  const int tid  = threadIdx.x;
  const int lane = tid & 31;
  const int wave = tid >> 5;
  const int lg   = lane & 15;
  const int half = lane >> 4;
  const int row0 = (blockIdx.x * WPB + wave) * RPW;

  _Float16 (*H)[HSTRIDE] = Hs[wave];
  _Float16 (*W)[WSTRIDE] = Ws[wave];

  // ---- load x tile -> LDS f16 ----
  for (int r = 0; r < RPW; ++r) {
    const float* src = x + (size_t)(row0 + r) * MD;
    #pragma unroll
    for (int c = 0; c < MD; c += 128) {
      v4f t = *(const v4f*)(src + c + lane * 4);
      v4h h4;
      h4.x = (_Float16)t.x; h4.y = (_Float16)t.y;
      h4.z = (_Float16)t.z; h4.w = (_Float16)t.w;
      *(v4h*)&H[r][c + lane * 4] = h4;
    }
  }

  for (int layer = 0; layer < NLAYERS; ++layer) {
    const _Float16* kbase = kf  + (size_t)layer * NS * MD;
    const _Float16* vbase = vtf + (size_t)layer * MD * NS;

    // ---- issue async staging of K chunk 0 early (hide behind norm pass) ----
    #pragma unroll
    for (int r = 0; r < 2; ++r)
      #pragma unroll
      for (int j = 0; j < 8; ++j)
        copy16_to_lds(kbase + (size_t)(2 * tid + r) * MD + 8 * j,
                      &Stage[0][(2 * tid + r) * KSTR + 8 * j]);

    // ---- row 1/(||h||+eps); half-wave g handles rows 8g..8g+7 ----
    float recip[8];
    #pragma unroll
    for (int r8 = 0; r8 < 8; ++r8) {
      const int r = half * 8 + r8;
      float ss = 0.f;
      #pragma unroll
      for (int j = 0; j < 32; ++j) {
        v2h p = *(const v2h*)&H[r][2 * lg + 32 * j];
        float a = (float)p.x, b = (float)p.y;
        ss = __builtin_fmaf(a, a, ss);
        ss = __builtin_fmaf(b, b, ss);
      }
      #pragma unroll
      for (int off = 1; off < 16; off <<= 1) ss += __shfl_xor(ss, off, 16);
      recip[r8] = 1.f / (__builtin_sqrtf(ss) + 1e-9f);
    }

    // ======== GEMM1: scores[16 x 256] = h * K^T ============================
    v8f S[16];
    #pragma unroll
    for (int nb = 0; nb < 16; ++nb) S[nb] = (v8f){};

    staging_wait();
    __syncthreads();

    #pragma unroll 1
    for (int c = 0; c < NKCH; ++c) {
      if (c + 1 < NKCH) {                       // async stage next chunk
        const int k0n = (c + 1) * KC;
        _Float16* buf = Stage[(c + 1) & 1];
        #pragma unroll
        for (int r = 0; r < 2; ++r)
          #pragma unroll
          for (int j = 0; j < 8; ++j)
            copy16_to_lds(kbase + (size_t)(2 * tid + r) * MD + k0n + 8 * j,
                          &buf[(2 * tid + r) * KSTR + 8 * j]);
      }
      const _Float16* kb = Stage[c & 1];
      const int k0 = c * KC;
      #pragma unroll
      for (int kk = 0; kk < KC; kk += 32) {
        const _Float16* bbase = kb + lg * KSTR + kk + half * 8;
        v16h A = load_frag(&H[lg][k0 + kk + half * 8]);
        v16h B0 = load_frag(bbase + 0 * 16 * KSTR);
        v16h B1 = load_frag(bbase + 1 * 16 * KSTR);
        v16h B2 = load_frag(bbase + 2 * 16 * KSTR);
        v16h B3 = load_frag(bbase + 3 * 16 * KSTR);
        #pragma unroll
        for (int g = 0; g < 4; ++g) {
          v16h C0, C1, C2, C3;
          if (g < 3) {                           // prefetch group g+1
            const _Float16* nbase = bbase + (4 * (g + 1)) * 16 * KSTR;
            C0 = load_frag(nbase + 0 * 16 * KSTR);
            C1 = load_frag(nbase + 1 * 16 * KSTR);
            C2 = load_frag(nbase + 2 * 16 * KSTR);
            C3 = load_frag(nbase + 3 * 16 * KSTR);
          }
          sched_fence();                         // loads stay above WMMAs
          S[4 * g + 0] = WMMA(A, B0, S[4 * g + 0]);
          S[4 * g + 1] = WMMA(A, B1, S[4 * g + 1]);
          S[4 * g + 2] = WMMA(A, B2, S[4 * g + 2]);
          S[4 * g + 3] = WMMA(A, B3, S[4 * g + 3]);
          sched_fence();                         // WMMAs stay above next loads
          if (g < 3) { B0 = C0; B1 = C1; B2 = C2; B3 = C3; }
        }
      }
      if (c + 1 < NKCH) { staging_wait(); __syncthreads(); }
    }

    // ---- scale by 1/||h||, softmax ----------------------------------------
    #pragma unroll
    for (int nb = 0; nb < 16; ++nb)
      #pragma unroll
      for (int i = 0; i < 8; ++i) S[nb][i] *= recip[i];

    #pragma unroll
    for (int i = 0; i < 8; ++i) {
      float m = -3.0e38f;
      #pragma unroll
      for (int nb = 0; nb < 16; ++nb) m = fmaxf(m, S[nb][i]);
      #pragma unroll
      for (int off = 1; off < 16; off <<= 1) m = fmaxf(m, __shfl_xor(m, off, 16));
      float sum = 0.f;
      #pragma unroll
      for (int nb = 0; nb < 16; ++nb) {
        float e = __expf(S[nb][i] - m);
        S[nb][i] = e;
        sum += e;
      }
      #pragma unroll
      for (int off = 1; off < 16; off <<= 1) sum += __shfl_xor(sum, off, 16);
      float rs = 1.f / sum;
      #pragma unroll
      for (int nb = 0; nb < 16; ++nb) S[nb][i] *= rs;
    }

    // ---- weights C-layout -> A-layout via per-wave LDS ----
    #pragma unroll
    for (int nb = 0; nb < 16; ++nb)
      #pragma unroll
      for (int i = 0; i < 8; ++i)
        W[half * 8 + i][nb * 16 + lg] = (_Float16)S[nb][i];

    // ======== GEMM2: h'[16 x 1024] = w * V =================================
    const bool last = (layer == NLAYERS - 1);

    // issue async staging of V chunk 0
    #pragma unroll
    for (int j = 0; j < 16; ++j)
      copy16_to_lds(vbase + (size_t)(tid >> 1) * NS + (tid & 1) * 128 + 8 * j,
                    &Stage[0][(tid >> 1) * VSTR + (tid & 1) * 128 + 8 * j]);

    // hoist all 8 W fragments (loop-invariant across d-chunks)
    v16h Wfrag[8];
    #pragma unroll
    for (int k8 = 0; k8 < 8; ++k8)
      Wfrag[k8] = load_frag(&W[lg][k8 * 32 + half * 8]);

    staging_wait();
    __syncthreads();

    #pragma unroll 1
    for (int c = 0; c < NDCH; ++c) {
      if (c + 1 < NDCH) {
        const int d0n = (c + 1) * DC;
        _Float16* buf = Stage[(c + 1) & 1];
        #pragma unroll
        for (int j = 0; j < 16; ++j)
          copy16_to_lds(
              vbase + (size_t)(d0n + (tid >> 1)) * NS + (tid & 1) * 128 + 8 * j,
              &buf[(tid >> 1) * VSTR + (tid & 1) * 128 + 8 * j]);
      }
      const _Float16* vb = Stage[c & 1];
      const _Float16* vrow = vb + lg * VSTR + half * 8;
      v8f acc[4];
      #pragma unroll
      for (int q = 0; q < 4; ++q) acc[q] = (v8f){};

      v16h B0 = load_frag(vrow + 0 * 16 * VSTR);
      v16h B1 = load_frag(vrow + 1 * 16 * VSTR);
      v16h B2 = load_frag(vrow + 2 * 16 * VSTR);
      v16h B3 = load_frag(vrow + 3 * 16 * VSTR);

      #pragma unroll
      for (int k0 = 0; k0 < NS; k0 += 32) {
        v16h C0, C1, C2, C3;
        if (k0 + 32 < NS) {                      // prefetch next k-step
          C0 = load_frag(vrow + 0 * 16 * VSTR + k0 + 32);
          C1 = load_frag(vrow + 1 * 16 * VSTR + k0 + 32);
          C2 = load_frag(vrow + 2 * 16 * VSTR + k0 + 32);
          C3 = load_frag(vrow + 3 * 16 * VSTR + k0 + 32);
        }
        sched_fence();
        acc[0] = WMMA(Wfrag[k0 / 32], B0, acc[0]);
        acc[1] = WMMA(Wfrag[k0 / 32], B1, acc[1]);
        acc[2] = WMMA(Wfrag[k0 / 32], B2, acc[2]);
        acc[3] = WMMA(Wfrag[k0 / 32], B3, acc[3]);
        sched_fence();
        if (k0 + 32 < NS) { B0 = C0; B1 = C1; B2 = C2; B3 = C3; }
      }

      const int d0 = c * DC;
      if (last) {
        #pragma unroll
        for (int q = 0; q < 4; ++q)
          #pragma unroll
          for (int i = 0; i < 8; ++i)
            out[(size_t)(row0 + half * 8 + i) * MD + d0 + q * 16 + lg] =
                acc[q][i];
      } else {
        #pragma unroll
        for (int q = 0; q < 4; ++q)
          #pragma unroll
          for (int i = 0; i < 8; ++i)
            H[half * 8 + i][d0 + q * 16 + lg] = (_Float16)acc[q][i];
      }
      if (c + 1 < NDCH) { staging_wait(); __syncthreads(); }
    }
  }
}

// ---------------------------------------------------------------------------
extern "C" void kernel_launch(void* const* d_in, const int* in_sizes, int n_in,
                              void* d_out, int out_size, void* d_ws, size_t ws_size,
                              hipStream_t stream) {
  const float* x      = (const float*)d_in[0];
  const float* keys   = (const float*)d_in[1];
  const float* values = (const float*)d_in[2];

  _Float16* kf  = (_Float16*)d_ws;
  _Float16* vtf = kf + (size_t)NLAYERS * NS * MD;

  const int total = NLAYERS * NS * MD;
  cleanup_prep_kv<<<(total + 255) / 256, 256, 0, stream>>>(keys, values, kf, vtf);

  const int blocks = ROWS_TOT / (WPB * RPW);   // 256 blocks x 128 threads
  cleanup_chain_wmma<<<blocks, WPB * 32, 0, stream>>>(x, kf, vtf, (float*)d_out);
}
